// HDCI_52458730553690
// MI455X (gfx1250) — compile-verified, matrix operand order checked by reading, not run on hardware
//
#include <hip/hip_runtime.h>
#include <math.h>

// ---------------------------------------------------------------------------
// MI455X (gfx1250) implementation of the hetero-GNN recommender forward pass.
// Dense node GEMMs -> v_wmma_f32_16x16x32_f16, B fragments register-resident,
// 4 M-tiles per block, float4 A loads, no divergent guards, branchless tanh.
// Edge segment ops -> wave-per-edge float4 gathers + f32 atomic scatters (L2).
// ---------------------------------------------------------------------------

#define D128 128

typedef __attribute__((ext_vector_type(16))) _Float16 v16h;
typedef __attribute__((ext_vector_type(8)))  float    v8f;

__device__ __forceinline__ float wave_sum(float v) {
#pragma unroll
  for (int off = 16; off > 0; off >>= 1) v += __shfl_xor(v, off, 32);
  return v;
}

// Branch-free tanh: tanh(x) = sign(x) * (1 - t) / (1 + t), t = exp(-2|x|).
// Saturates naturally for large |x| (t -> 0); no range-split control flow.
__device__ __forceinline__ float tanh_fast(float x) {
  const float ax = __builtin_fabsf(x);
  const float t = __expf(-2.0f * ax);
  const float y = (1.0f - t) / (1.0f + t);
  return __builtin_copysignf(y, x);
}

// ---------------------------------------------------------------------------
// C[N,128] = act( [A0 | A1][N, KTOT] @ W[KTOT, 128] + bias )
// KTOT = 128 (A1 unused, pass A0) or 256 (concat of two [N,128] blocks).
// ACT: 0 = none, 1 = tanh.  Requires N % 16 == 0 (holds: N = 50000).
// One wave computes one 16x16 tile; 8 waves/block cover Dout=128; each block
// iterates MT m-tiles reusing register-resident B fragments.
// CDNA5 16-bit WMMA lane layout:
//   A: lane l (l<16) holds M=tileM+l, K = kb..kb+7, kb+16..kb+23 (kb+=8 for
//      lanes 16..31).  B symmetric with N instead of M.
//   C: vgpr r -> row tileM + r + (lane>=16 ? 8 : 0), col tileN + (lane&15).
// ---------------------------------------------------------------------------
template <int KTOT, int MT, int ACT>
__global__ __launch_bounds__(256) void k_gemm_wmma(
    const float* __restrict__ A0,    // [N,128]
    const float* __restrict__ A1,    // [N,128] (= A0 when KTOT==128)
    const float* __restrict__ W,     // [KTOT,128]
    const float* __restrict__ bias,  // nullptr -> no bias
    float* __restrict__ out, int N) {
  constexpr int NSTEP = KTOT / 32;
  const int lane = threadIdx.x & 31;
  const int wave = threadIdx.x >> 5;
  const int half = lane >> 4;
  const int lr   = lane & 15;
  const int col  = (wave << 4) + lr;

  // Preload B fragments for this column tile (once per block, reused MT times)
  v16h b[NSTEP];
#pragma unroll
  for (int s = 0; s < NSTEP; ++s) {
    const int kb = s * 32 + (half ? 8 : 0);
#pragma unroll
    for (int i = 0; i < 16; ++i) {
      const int kk = kb + i + ((i < 8) ? 0 : 8);
      b[s][i] = (_Float16)W[kk * D128 + col];
    }
  }

  const float bv = bias ? bias[col] : 0.0f;

#pragma unroll 1
  for (int m = 0; m < MT; ++m) {
    const int tileM = (blockIdx.x * MT + m) << 4;
    if (tileM >= N) return;                       // uniform per wave
    const int row = tileM + lr;
    const float* __restrict__ r0 = A0 + row * D128;
    const float* __restrict__ r1 = A1 + row * D128;
    v8f c = {};
#pragma unroll
    for (int s = 0; s < NSTEP; ++s) {
      const int k0 = s * 32;
      const float* __restrict__ rp = (k0 < D128) ? r0 : r1;  // compile-time per s
      const int kb = ((k0 < D128) ? k0 : (k0 - D128)) + (half ? 8 : 0);
      const float4 f0 = *(const float4*)(rp + kb);
      const float4 f1 = *(const float4*)(rp + kb + 4);
      const float4 f2 = *(const float4*)(rp + kb + 16);
      const float4 f3 = *(const float4*)(rp + kb + 20);
      v16h a;
      a[0]  = (_Float16)f0.x; a[1]  = (_Float16)f0.y;
      a[2]  = (_Float16)f0.z; a[3]  = (_Float16)f0.w;
      a[4]  = (_Float16)f1.x; a[5]  = (_Float16)f1.y;
      a[6]  = (_Float16)f1.z; a[7]  = (_Float16)f1.w;
      a[8]  = (_Float16)f2.x; a[9]  = (_Float16)f2.y;
      a[10] = (_Float16)f2.z; a[11] = (_Float16)f2.w;
      a[12] = (_Float16)f3.x; a[13] = (_Float16)f3.y;
      a[14] = (_Float16)f3.z; a[15] = (_Float16)f3.w;
      c = __builtin_amdgcn_wmma_f32_16x16x32_f16(false, a, false, b[s],
                                                 (short)0, c, false, false);
    }
#pragma unroll
    for (int r = 0; r < 8; ++r) {
      const int orow = tileM + r + (half ? 8 : 0);
      float v = c[r] + bv;
      if (ACT == 1) v = tanh_fast(v);
      out[orow * D128 + col] = v;
    }
  }
}

// ---------------------------------------------------------------------------
// Edge attention: a[e] = exp( sum_d th[src,d] * hdst[dst,d] * wA[d] ),
// att[dst] += a[e].   One wave32 per edge, float4 per lane (128 = 32*4).
// ---------------------------------------------------------------------------
__global__ __launch_bounds__(256) void k_edge_att(
    const float* __restrict__ th, const float* __restrict__ hdst,
    const float* __restrict__ wA,
    const int* __restrict__ src, const int* __restrict__ dst,
    float* __restrict__ a_e, float* __restrict__ att, int E) {
  const int lane = threadIdx.x & 31;
  const int e = blockIdx.x * 8 + (threadIdx.x >> 5);
  if (e >= E) return;
  const int s = src[e], d = dst[e];
  const float4 t = *(const float4*)(th   + (s << 7) + lane * 4);
  const float4 h = *(const float4*)(hdst + (d << 7) + lane * 4);
  const float4 w = *(const float4*)(wA + lane * 4);
  float p = t.x * h.x * w.x + t.y * h.y * w.y + t.z * h.z * w.z + t.w * h.w * w.w;
  p = wave_sum(p);
  if (lane == 0) {
    const float a = __expf(p);
    a_e[e] = a;
    atomicAdd(att + d, a);
  }
}

// out[dst] += hsrc[src] * (a[e] / (att[dst] + eps))
__global__ __launch_bounds__(256) void k_edge_agg(
    const float* __restrict__ hsrc,
    const int* __restrict__ src, const int* __restrict__ dst,
    const float* __restrict__ a_e, const float* __restrict__ att,
    float* __restrict__ out, int E) {
  const int lane = threadIdx.x & 31;
  const int e = blockIdx.x * 8 + (threadIdx.x >> 5);
  if (e >= E) return;
  const int s = src[e], d = dst[e];
  float w = 0.0f;
  if (lane == 0) w = a_e[e] / (att[d] + 1e-12f);
  w = __shfl(w, 0, 32);
  const float4 h = *(const float4*)(hsrc + (s << 7) + lane * 4);
  float* o = out + (d << 7) + lane * 4;
  atomicAdd(o + 0, h.x * w);
  atomicAdd(o + 1, h.y * w);
  atomicAdd(o + 2, h.z * w);
  atomicAdd(o + 3, h.w * w);
}

// degree accumulation for GraphConv(norm='both')
__global__ void k_deg(const int* __restrict__ src, const int* __restrict__ dst,
                      float* __restrict__ dego, float* __restrict__ degi, int E) {
  const int e = blockIdx.x * blockDim.x + threadIdx.x;
  if (e >= E) return;
  atomicAdd(dego + src[e], 1.0f);
  atomicAdd(degi + dst[e], 1.0f);
}

// agg[dst] += h[src] * rsqrt(max(deg_out[src],1))
__global__ __launch_bounds__(256) void k_gconv_scatter(
    const float* __restrict__ h, const float* __restrict__ dego,
    const int* __restrict__ src, const int* __restrict__ dst,
    float* __restrict__ agg, int E) {
  const int lane = threadIdx.x & 31;
  const int e = blockIdx.x * 8 + (threadIdx.x >> 5);
  if (e >= E) return;
  const int s = src[e], d = dst[e];
  float rs = 0.0f;
  if (lane == 0) rs = rsqrtf(fmaxf(dego[s], 1.0f));
  rs = __shfl(rs, 0, 32);
  const float4 hv = *(const float4*)(h + (s << 7) + lane * 4);
  float* o = agg + (d << 7) + lane * 4;
  atomicAdd(o + 0, hv.x * rs);
  atomicAdd(o + 1, hv.y * rs);
  atomicAdd(o + 2, hv.z * rs);
  atomicAdd(o + 3, hv.w * rs);
}

// z *= rsqrt(max(deg_in[node],1))
__global__ void k_gconv_scale(float* __restrict__ z,
                              const float* __restrict__ degi, int ND) {
  const int i = blockIdx.x * blockDim.x + threadIdx.x;
  if (i >= ND) return;
  z[i] *= rsqrtf(fmaxf(degi[i >> 7], 1.0f));
}

// x = layernorm(relu(x)) * g + b    (row = 128, one wave per row)
__global__ __launch_bounds__(256) void k_relu_ln(
    float* __restrict__ x, const float* __restrict__ g,
    const float* __restrict__ b, int N) {
  const int lane = threadIdx.x & 31;
  const int row = blockIdx.x * 8 + (threadIdx.x >> 5);
  if (row >= N) return;
  float4 v = *(const float4*)(x + (row << 7) + lane * 4);
  v.x = fmaxf(v.x, 0.0f); v.y = fmaxf(v.y, 0.0f);
  v.z = fmaxf(v.z, 0.0f); v.w = fmaxf(v.w, 0.0f);
  const float m = wave_sum(v.x + v.y + v.z + v.w) * (1.0f / 128.0f);
  const float dx = v.x - m, dy = v.y - m, dz = v.z - m, dw = v.w - m;
  const float var = wave_sum(dx * dx + dy * dy + dz * dz + dw * dw) * (1.0f / 128.0f);
  const float inv = rsqrtf(var + 1e-5f);
  const float4 gg = *(const float4*)(g + lane * 4);
  const float4 bb = *(const float4*)(b + lane * 4);
  float4 o;
  o.x = dx * inv * gg.x + bb.x;
  o.y = dy * inv * gg.y + bb.y;
  o.z = dz * inv * gg.z + bb.z;
  o.w = dw * inv * gg.w + bb.w;
  *(float4*)(x + (row << 7) + lane * 4) = o;
}

// acc += sum_n dot(t[n,:], w2)   (semantic-attention logits, summed over N)
__global__ __launch_bounds__(256) void k_dot_accum(
    const float* __restrict__ t, const float* __restrict__ w2,
    float* __restrict__ acc, int N) {
  const int lane = threadIdx.x & 31;
  const int row = blockIdx.x * 8 + (threadIdx.x >> 5);
  if (row >= N) return;
  const float4 v = *(const float4*)(t + (row << 7) + lane * 4);
  const float4 w = *(const float4*)(w2 + lane * 4);
  const float p = wave_sum(v.x * w.x + v.y * w.y + v.z * w.z + v.w * w.w);
  if (lane == 0) atomicAdd(acc, p);
}

// beta = softmax([ssum0/N, ssum1/N])
__global__ void k_beta(const float* __restrict__ ssum,
                       float* __restrict__ beta, float invN) {
  if (threadIdx.x == 0 && blockIdx.x == 0) {
    const float m0 = ssum[0] * invN, m1 = ssum[1] * invN;
    const float mx = fmaxf(m0, m1);
    const float e0 = __expf(m0 - mx), e1 = __expf(m1 - mx);
    const float inv = 1.0f / (e0 + e1);
    beta[0] = e0 * inv;
    beta[1] = e1 * inv;
  }
}

// h2 = beta[0]*z0 + beta[1]*z1
__global__ void k_combine(const float* __restrict__ z0,
                          const float* __restrict__ z1,
                          const float* __restrict__ beta,
                          float* __restrict__ h2, int ND) {
  const int i = blockIdx.x * blockDim.x + threadIdx.x;
  if (i >= ND) return;
  h2[i] = beta[0] * z0[i] + beta[1] * z1[i];
}

// out[b, :] = src[idx[b], :]
__global__ void k_gather(const float* __restrict__ src,
                         const int* __restrict__ idx,
                         float* __restrict__ out, int B) {
  const int i = blockIdx.x * blockDim.x + threadIdx.x;
  if (i >= (B << 7)) return;
  const int b = i >> 7, d = i & 127;
  out[i] = src[(idx[b] << 7) + d];
}

// ---------------------------------------------------------------------------
extern "C" void kernel_launch(void* const* d_in, const int* in_sizes, int n_in,
                              void* d_out, int out_size, void* d_ws, size_t ws_size,
                              hipStream_t stream) {
  const float* feat_user = (const float*)d_in[0];
  const float* feat_item = (const float*)d_in[1];
  const float* W_T_ui = (const float*)d_in[2];
  const float* w_A_ui = (const float*)d_in[3];
  const float* W_T_iu = (const float*)d_in[4];
  const float* w_A_iu = (const float*)d_in[5];
  const float* ln1_g = (const float*)d_in[6];
  const float* ln1_b = (const float*)d_in[7];
  const float* sa_u_W1 = (const float*)d_in[8];
  const float* sa_u_b1 = (const float*)d_in[9];
  const float* sa_u_w2 = (const float*)d_in[10];
  const float* sa_i_W1 = (const float*)d_in[11];
  const float* sa_i_b1 = (const float*)d_in[12];
  const float* sa_i_w2 = (const float*)d_in[13];
  const float* ul1_W = (const float*)d_in[14];
  const float* ul1_b = (const float*)d_in[15];
  const float* ul2_W = (const float*)d_in[16];
  const float* ul2_b = (const float*)d_in[17];
  const float* il1_W = (const float*)d_in[18];
  const float* il1_b = (const float*)d_in[19];
  const float* il2_W = (const float*)d_in[20];
  const float* il2_b = (const float*)d_in[21];
  const float* ln2_g = (const float*)d_in[22];
  const float* ln2_b = (const float*)d_in[23];
  const int* ui_src = (const int*)d_in[24];
  const int* ui_dst = (const int*)d_in[25];
  const int* iu_src = (const int*)d_in[26];
  const int* iu_dst = (const int*)d_in[27];
  const int* mp_u1_src = (const int*)d_in[28];
  const int* mp_u1_dst = (const int*)d_in[29];
  const int* mp_u2_src = (const int*)d_in[30];
  const int* mp_u2_dst = (const int*)d_in[31];
  const int* mp_i1_src = (const int*)d_in[32];
  const int* mp_i1_dst = (const int*)d_in[33];
  const int* mp_i2_src = (const int*)d_in[34];
  const int* mp_i2_dst = (const int*)d_in[35];
  const int* user_idx = (const int*)d_in[36];
  const int* item_idx = (const int*)d_in[37];
  const int* neg_item_idx = (const int*)d_in[38];

  const int U = in_sizes[0] / D128;
  const int I = in_sizes[1] / D128;
  const int E = in_sizes[24];
  const int B = in_sizes[36];
  const int Nmax = (U > I) ? U : I;
  const size_t NDu = (size_t)U * D128;
  const size_t NDi = (size_t)I * D128;
  const size_t NDmax = (NDu > NDi) ? NDu : NDi;

  constexpr int MT = 4;  // M-tiles per GEMM block
  auto gemmGrid = [](int N) { return ((N + 15) / 16 + MT - 1) / MT; };

  // ---- workspace carve-out (floats) ----
  float* ws   = (float*)d_ws;
  float* th   = ws + 0 * NDmax;   // projected features (t_h), reused per etype
  float* h1u  = ws + 1 * NDmax;
  float* h1i  = ws + 2 * NDmax;
  float* z0   = ws + 3 * NDmax;   // GraphConv outputs (reused user->item)
  float* z1   = ws + 4 * NDmax;
  float* h2u  = ws + 5 * NDmax;
  float* h2i  = ws + 6 * NDmax;
  float* tmp  = ws + 7 * NDmax;   // tanh-GEMM / fusion-layer1 scratch
  float* ue   = ws + 8 * NDmax;
  float* ie   = ws + 9 * NDmax;
  float* a_e  = ws + 10 * NDmax;            // [E]
  float* att  = a_e + E;                    // [Nmax]
  float* dego = att + Nmax;                 // [Nmax]
  float* degi = dego + Nmax;                // [Nmax]
  float* ssum = degi + Nmax;                // [2]
  float* betp = ssum + 2;                   // [2]

  const int edgeBlocks = (E + 7) / 8;
  const int edgeTpb    = (E + 255) / 256;

  // ======================= RelationalAGG (h1) =============================
  // h1_item: users --(ui)--> items
  hipMemsetAsync(att, 0, (size_t)I * sizeof(float), stream);
  hipMemsetAsync(h1i, 0, NDi * sizeof(float), stream);
  k_gemm_wmma<128, MT, 0><<<gemmGrid(U), 256, 0, stream>>>(
      feat_user, feat_user, W_T_ui, nullptr, th, U);
  k_edge_att<<<edgeBlocks, 256, 0, stream>>>(th, feat_item, w_A_ui,
                                             ui_src, ui_dst, a_e, att, E);
  k_edge_agg<<<edgeBlocks, 256, 0, stream>>>(feat_user, ui_src, ui_dst,
                                             a_e, att, h1i, E);
  k_relu_ln<<<(I + 7) / 8, 256, 0, stream>>>(h1i, ln1_g, ln1_b, I);

  // h1_user: items --(iu)--> users
  hipMemsetAsync(att, 0, (size_t)U * sizeof(float), stream);
  hipMemsetAsync(h1u, 0, NDu * sizeof(float), stream);
  k_gemm_wmma<128, MT, 0><<<gemmGrid(I), 256, 0, stream>>>(
      feat_item, feat_item, W_T_iu, nullptr, th, I);
  k_edge_att<<<edgeBlocks, 256, 0, stream>>>(th, feat_user, w_A_iu,
                                             iu_src, iu_dst, a_e, att, E);
  k_edge_agg<<<edgeBlocks, 256, 0, stream>>>(feat_item, iu_src, iu_dst,
                                             a_e, att, h1u, E);
  k_relu_ln<<<(U + 7) / 8, 256, 0, stream>>>(h1u, ln1_g, ln1_b, U);

  // ======================= HAN layer (h2) =================================
  auto gconv = [&](const float* feat, const int* s, const int* d,
                   int N, size_t NDn, float* z) {
    hipMemsetAsync(dego, 0, (size_t)N * sizeof(float), stream);
    hipMemsetAsync(degi, 0, (size_t)N * sizeof(float), stream);
    hipMemsetAsync(z, 0, NDn * sizeof(float), stream);
    k_deg<<<edgeTpb, 256, 0, stream>>>(s, d, dego, degi, E);
    k_gconv_scatter<<<edgeBlocks, 256, 0, stream>>>(feat, dego, s, d, z, E);
    k_gconv_scale<<<(int)((NDn + 255) / 256), 256, 0, stream>>>(z, degi, (int)NDn);
  };
  auto semantic = [&](const float* za, const float* zb, const float* W1,
                      const float* b1, const float* w2, int N, size_t NDn,
                      float* h2) {
    hipMemsetAsync(ssum, 0, 2 * sizeof(float), stream);
    k_gemm_wmma<128, MT, 1><<<gemmGrid(N), 256, 0, stream>>>(
        za, za, W1, b1, tmp, N);
    k_dot_accum<<<(N + 7) / 8, 256, 0, stream>>>(tmp, w2, ssum + 0, N);
    k_gemm_wmma<128, MT, 1><<<gemmGrid(N), 256, 0, stream>>>(
        zb, zb, W1, b1, tmp, N);
    k_dot_accum<<<(N + 7) / 8, 256, 0, stream>>>(tmp, w2, ssum + 1, N);
    k_beta<<<1, 32, 0, stream>>>(ssum, betp, 1.0f / (float)N);
    k_combine<<<(int)((NDn + 255) / 256), 256, 0, stream>>>(za, zb, betp, h2,
                                                            (int)NDn);
  };

  gconv(feat_user, mp_u1_src, mp_u1_dst, U, NDu, z0);
  gconv(feat_user, mp_u2_src, mp_u2_dst, U, NDu, z1);
  semantic(z0, z1, sa_u_W1, sa_u_b1, sa_u_w2, U, NDu, h2u);

  gconv(feat_item, mp_i1_src, mp_i1_dst, I, NDi, z0);
  gconv(feat_item, mp_i2_src, mp_i2_dst, I, NDi, z1);
  semantic(z0, z1, sa_i_W1, sa_i_b1, sa_i_w2, I, NDi, h2i);

  // ======================= fusion layers ==================================
  auto fuse = [&](const float* h1, const float* h2, const float* feat,
                  const float* L1W, const float* L1b, const float* L2W,
                  const float* L2b, int N, float* outbuf) {
    k_gemm_wmma<256, MT, 0><<<gemmGrid(N), 256, 0, stream>>>(
        h1, h2, L1W, L1b, tmp, N);
    k_gemm_wmma<256, MT, 0><<<gemmGrid(N), 256, 0, stream>>>(
        tmp, feat, L2W, L2b, outbuf, N);
    k_relu_ln<<<(N + 7) / 8, 256, 0, stream>>>(outbuf, ln2_g, ln2_b, N);
  };
  fuse(h1u, h2u, feat_user, ul1_W, ul1_b, ul2_W, ul2_b, U, ue);
  fuse(h1i, h2i, feat_item, il1_W, il1_b, il2_W, il2_b, I, ie);

  // ======================= batched gathers ================================
  float* out = (float*)d_out;
  const int GD = (B * D128 + 255) / 256;
  k_gather<<<GD, 256, 0, stream>>>(ue, user_idx, out, B);
  k_gather<<<GD, 256, 0, stream>>>(ie, item_idx, out + (size_t)B * D128, B);
  k_gather<<<GD, 256, 0, stream>>>(ie, neg_item_idx, out + 2 * (size_t)B * D128, B);
}
